// BabyRadixAttention_81174881894411
// MI455X (gfx1250) — matrix operand order repaired and checked by reading, not compiled, not run on hardware
//
#include <hip/hip_runtime.h>
#include <hip/hip_bf16.h>

// ---------------------------------------------------------------------------
// Flash-attention (causal, GQA) for MI455X / gfx1250 using bf16 WMMA.
//   q: [B*S, H, D] f32   k,v: [B*S, HK, D] f32   out: [B*S, H*D] f32
//
// Pass 1: one-time K/V f32->bf16 conversion into d_ws:
//   kbf: [B, HK, S, D]  (32-key block = contiguous 8KB)
//   vbf: [B, HK, D, S]  (transposed: LDS [d][k] tile = 128 contiguous rows)
// Pass 2: attention. K/V double-buffered in LDS via GLOBAL_LOAD_ASYNC_TO_LDS
// (ASYNCcnt): block kb+1 streams in while block kb computes. One workgroup =
// 4 waves = the 4 query heads sharing one kv head; each wave owns a 16-row
// query tile. Causal mask only instantiated for the final key block.
// Softmax row reductions: single-instruction DPP row_ror combine ops.
// ---------------------------------------------------------------------------

constexpr int Bc = 4;      // batch
constexpr int Sc = 2048;   // seq len
constexpr int Hc = 16;     // q heads
constexpr int HKc = 4;     // kv heads
constexpr int Dc = 128;    // head dim
constexpr int REP = Hc / HKc;   // 4 q heads per kv head
constexpr int QT = 16;     // query rows per wave
constexpr int KT = 32;     // keys per block

constexpr int K_STRIDE = Dc + 8;  // bf16 units, padded (136 elem = 272B, 16B-mult)
constexpr int V_STRIDE = KT + 8;  // V transposed [d][k], stride 40 elem = 80B (8B-mult)
constexpr int P_STRIDE = KT + 8;  // per-wave P scratch stride 40

// SCALE * log2(e), folded into Q at load so softmax can use exp2f directly.
constexpr float SCALE_LOG2E = 0.08838834764831845f * 1.4426950408889634f;

typedef __attribute__((ext_vector_type(16))) __bf16 v16bf;
typedef __attribute__((ext_vector_type(8)))  float  v8f;

union Frag {
    unsigned int u[8];
    v16bf bf;
};

__device__ __forceinline__ unsigned short f2bf(float x) {
    unsigned int u = __float_as_uint(x);
    unsigned int r = u + 0x7FFFu + ((u >> 16) & 1u);   // round-to-nearest-even
    return (unsigned short)(r >> 16);
}
__device__ __forceinline__ unsigned int pack2bf(float lo, float hi) {
    return (unsigned int)f2bf(lo) | ((unsigned int)f2bf(hi) << 16);
}
__device__ __forceinline__ v8f vzero8() {
    v8f r;
#pragma unroll
    for (int i = 0; i < 8; ++i) r[i] = 0.0f;
    return r;
}

// --- all-reduce within each 16-lane group: one DPP-combine op per step ------
// (EXEC is all-1s at these points, so every row_ror source lane is valid.)
__device__ __forceinline__ float rowmax16(float x) {
    asm("v_max_num_f32_dpp %0, %0, %0 row_ror:1 row_mask:0xf bank_mask:0xf" : "+v"(x));
    asm("v_max_num_f32_dpp %0, %0, %0 row_ror:2 row_mask:0xf bank_mask:0xf" : "+v"(x));
    asm("v_max_num_f32_dpp %0, %0, %0 row_ror:4 row_mask:0xf bank_mask:0xf" : "+v"(x));
    asm("v_max_num_f32_dpp %0, %0, %0 row_ror:8 row_mask:0xf bank_mask:0xf" : "+v"(x));
    return x;
}
__device__ __forceinline__ float rowsum16(float x) {
    asm("v_add_f32_dpp %0, %0, %0 row_ror:1 row_mask:0xf bank_mask:0xf" : "+v"(x));
    asm("v_add_f32_dpp %0, %0, %0 row_ror:2 row_mask:0xf bank_mask:0xf" : "+v"(x));
    asm("v_add_f32_dpp %0, %0, %0 row_ror:4 row_mask:0xf bank_mask:0xf" : "+v"(x));
    asm("v_add_f32_dpp %0, %0, %0 row_ror:8 row_mask:0xf bank_mask:0xf" : "+v"(x));
    return x;
}

// ---------------------------------------------------------------------------
// Pass 1: K/V -> bf16 workspace (K head-major, V head-major + transposed).
// grid = B*HK*(S/64), 256 threads.
// ---------------------------------------------------------------------------
__global__ __launch_bounds__(256)
void kv_to_bf16(const float* __restrict__ k, const float* __restrict__ v,
                unsigned short* __restrict__ kbf, unsigned short* __restrict__ vbf) {
    __shared__ unsigned short t_lds[Dc * 64];   // [d][s] transpose tile, 16KB

    const int wg   = blockIdx.x;
    const int sblk = wg % (Sc / 64);
    const int kvh  = (wg / (Sc / 64)) % HKc;
    const int b    = wg / ((Sc / 64) * HKc);
    const int s0   = sblk * 64;
    const int tid  = threadIdx.x;

#pragma unroll
    for (int i = 0; i < 8; ++i) {
        int f  = i * 256 + tid;          // float4 id, 0..2047 (64 rows x 32)
        int r  = f >> 5;                 // s row within tile
        int c4 = f & 31;                 // float4 along d
        size_t gsrc = ((size_t)(b * Sc + s0 + r) * HKc + kvh) * Dc + c4 * 4;
        float4 kd = *(const float4*)(k + gsrc);
        uint2 kp;
        kp.x = pack2bf(kd.x, kd.y);
        kp.y = pack2bf(kd.z, kd.w);
        *(uint2*)(kbf + ((size_t)(b * HKc + kvh) * Sc + s0 + r) * Dc + c4 * 4) = kp;
        float4 vd = *(const float4*)(v + gsrc);
        t_lds[(c4 * 4 + 0) * 64 + r] = f2bf(vd.x);
        t_lds[(c4 * 4 + 1) * 64 + r] = f2bf(vd.y);
        t_lds[(c4 * 4 + 2) * 64 + r] = f2bf(vd.z);
        t_lds[(c4 * 4 + 3) * 64 + r] = f2bf(vd.w);
    }
    __syncthreads();
    // write transposed V: 128 d-rows x 64 s (128B per row); thread = half row
    const int d  = tid >> 1;
    const int h2 = tid & 1;
    uint4 val = *(const uint4*)(&t_lds[d * 64 + h2 * 32]);
    *(uint4*)(vbf + ((size_t)(b * HKc + kvh) * Dc + d) * Sc + s0 + h2 * 32) = val;
}

// ---------------------------------------------------------------------------
// One 32-key block of flash attention (MASK = causal masking for last block).
// ---------------------------------------------------------------------------
template<bool MASK>
__device__ __forceinline__ void attn_block(
    const unsigned short* __restrict__ kbuf,   // LDS K tile [32][K_STRIDE]
    const unsigned short* __restrict__ vbuf,   // LDS V tile [128][V_STRIDE]
    unsigned short* __restrict__ p_lds,        // P scratch (all waves)
    const Frag (&qa)[4], v8f (&oacc)[8], float (&mrow)[8], float (&lrow)[8],
    int kstart, int q0, int half, int ln, int wave) {

    // --- scores: S = Q @ K^T, two 16x16 tiles, chained over D -----------------
    v8f s0 = vzero8(), s1 = vzero8();
#pragma unroll
    for (int c = 0; c < 4; ++c) {
        Frag kb0, kb1;
#pragma unroll
        for (int vv = 0; vv < 8; ++vv) {
            int d = c * 32 + 2 * vv + (vv >= 4 ? 8 : 0) + (half ? 8 : 0);   // B-frag pair (d, d+1)
            kb0.u[vv] = *(const unsigned int*)(&kbuf[(ln)      * K_STRIDE + d]);
            kb1.u[vv] = *(const unsigned int*)(&kbuf[(16 + ln) * K_STRIDE + d]);
        }
        s0 = __builtin_amdgcn_wmma_f32_16x16x32_bf16(false, qa[c].bf, false, kb0.bf,
                                                     (short)0, s0, false, false);
        s1 = __builtin_amdgcn_wmma_f32_16x16x32_bf16(false, qa[c].bf, false, kb1.bf,
                                                     (short)0, s1, false, false);
    }

    // --- (optional) causal mask + online softmax (DPP reductions) -------------
    float rscale[8];
#pragma unroll
    for (int vv = 0; vv < 8; ++vv) {
        const int M = vv + (half ? 8 : 0);
        float a0 = s0[vv], a1 = s1[vv];
        if (MASK) {
            const int qg = q0 + M;
            if (kstart + ln > qg)      a0 = -1e30f;
            if (kstart + 16 + ln > qg) a1 = -1e30f;
        }
        const float mx = rowmax16(fmaxf(a0, a1));
        const float mn = fmaxf(mrow[vv], mx);
        const float sc = exp2f(mrow[vv] - mn);
        mrow[vv] = mn;
        const float p0 = exp2f(a0 - mn);
        const float p1 = exp2f(a1 - mn);
        lrow[vv] = lrow[vv] * sc + rowsum16(p0 + p1);
        rscale[vv] = sc;
        // bounce P through per-wave LDS: C-layout -> A-layout
        const int pbase = (wave * QT + M) * P_STRIDE;
        p_lds[pbase + ln]      = f2bf(p0);
        p_lds[pbase + 16 + ln] = f2bf(p1);
    }

#pragma unroll
    for (int t = 0; t < 8; ++t)
#pragma unroll
        for (int vv = 0; vv < 8; ++vv) oacc[t][vv] *= rscale[vv];

    // --- P A-fragment (same-wave LDS ops are in-order) ------------------------
    Frag pa;
#pragma unroll
    for (int vv = 0; vv < 8; ++vv) {
        int kk = 2 * vv + (vv >= 4 ? 8 : 0) + (half ? 8 : 0);
        pa.u[vv] = *(const unsigned int*)(&p_lds[(wave * QT + ln) * P_STRIDE + kk]);
    }

    // --- O += P @ V : 8 output tiles along D ----------------------------------
#pragma unroll
    for (int t = 0; t < 8; ++t) {
        Frag vb;
#pragma unroll
        for (int vv = 0; vv < 8; ++vv) {
            int kk = 2 * vv + (vv >= 4 ? 8 : 0) + (half ? 8 : 0);
            vb.u[vv] = *(const unsigned int*)(&vbuf[(t * 16 + ln) * V_STRIDE + kk]);
        }
        oacc[t] = __builtin_amdgcn_wmma_f32_16x16x32_bf16(false, pa.bf, false, vb.bf,
                                                          (short)0, oacc[t], false, false);
    }
}

// ---------------------------------------------------------------------------
// Pass 2: attention from bf16 workspace, double-buffered async-to-LDS staging.
// ---------------------------------------------------------------------------
__global__ __launch_bounds__(128)
void attn_fwd_bf16ws(const unsigned short* __restrict__ kbf,
                     const unsigned short* __restrict__ vbf,
                     const float* __restrict__ q,
                     float* __restrict__ out) {
    __shared__ unsigned short k_lds[2][KT * K_STRIDE];   // 2 x 8704 B
    __shared__ unsigned short v_lds[2][Dc * V_STRIDE];   // 2 x 10240 B
    __shared__ unsigned short p_lds[4 * QT * P_STRIDE];  // 5120 B

    const int tid  = threadIdx.x;
    const int wave = tid >> 5;
    const int lane = tid & 31;
    const int half = lane >> 4;
    const int ln   = lane & 15;

    const int wg    = blockIdx.x;
    const int qtile = wg % (Sc / QT);
    const int kvh   = (wg / (Sc / QT)) % HKc;
    const int bidx  = wg / ((Sc / QT) * HKc);
    const int q0    = qtile * QT;
    const int h     = kvh * REP + wave;
    const int bhk   = bidx * HKc + kvh;

    // --- Q tile (16 x 128) -> 4 bf16 A-fragments (pre-scaled) ------------------
    Frag qa[4];
    {
        const float* qrow = q + (((size_t)(bidx * Sc + q0 + ln) * Hc + h) * Dc);
#pragma unroll
        for (int c = 0; c < 4; ++c) {
#pragma unroll
            for (int vv = 0; vv < 8; ++vv) {
                int d = c * 32 + 2 * vv + (vv >= 4 ? 8 : 0) + (half ? 8 : 0);
                float2 f = *(const float2*)(qrow + d);
                qa[c].u[vv] = pack2bf(f.x * SCALE_LOG2E, f.y * SCALE_LOG2E);
            }
        }
    }

    v8f oacc[8];
#pragma unroll
    for (int t = 0; t < 8; ++t) oacc[t] = vzero8();
    float mrow[8], lrow[8];
#pragma unroll
    for (int vv = 0; vv < 8; ++vv) { mrow[vv] = -1e30f; lrow[vv] = 0.0f; }

    // LDS byte offsets (low 32 bits of flat shared address == LDS offset)
    const unsigned klds0 = (unsigned)(unsigned long long)&k_lds[0][0];
    const unsigned vlds0 = (unsigned)(unsigned long long)&v_lds[0][0];
    const unsigned long long kga = (unsigned long long)(kbf + (size_t)bhk * Sc * Dc);
    const unsigned long long vga = (unsigned long long)(vbf + (size_t)bhk * Dc * Sc);

    // issue async copies of 32-key block at `kstart` into LDS buffer `buf`
    auto issue_stage = [&](int buf, int kstart) {
        const unsigned kldsb = klds0 + (unsigned)(buf * (KT * K_STRIDE * 2));
        const unsigned vldsb = vlds0 + (unsigned)(buf * (Dc * V_STRIDE * 2));
        unsigned long long kbase = kga + (unsigned long long)kstart * Dc * 2;
#pragma unroll
        for (int i = 0; i < 4; ++i) {
            int cc = i * 128 + tid;          // 16B chunk id, 0..511
            int r  = cc >> 4;                // key row
            int c  = cc & 15;                // 16B chunk within row
            unsigned lds = kldsb + (unsigned)(r * (K_STRIDE * 2) + c * 16);
            unsigned off = (unsigned)(r * (Dc * 2) + c * 16);
            asm volatile("global_load_async_to_lds_b128 %0, %1, %2"
                         :: "v"(lds), "v"(off), "s"(kbase) : "memory");
        }
        unsigned long long vbase = vga + (unsigned long long)kstart * 2;
#pragma unroll
        for (int i = 0; i < 8; ++i) {
            int cc = i * 128 + tid;          // 8B chunk id, 0..1023
            int d  = cc >> 3;                // d row
            int c  = cc & 7;                 // 8B chunk within row
            unsigned lds = vldsb + (unsigned)(d * (V_STRIDE * 2) + c * 8);
            unsigned off = (unsigned)(d * (Sc * 2) + c * 8);
            asm volatile("global_load_async_to_lds_b64 %0, %1, %2"
                         :: "v"(lds), "v"(off), "s"(vbase) : "memory");
        }
    };

    const int nkb = (q0 >> 5) + 1;   // key blocks (only last needs the mask)

    issue_stage(0, 0);               // prologue: prefetch block 0

    int kb = 0;
    for (; kb < nkb - 1; ++kb) {
        asm volatile("s_wait_asynccnt 0x0" ::: "memory");  // my buf[kb] loads done
        __syncthreads();             // all waves: data ready, old buffer drained
        issue_stage((kb + 1) & 1, (kb + 1) * KT);          // overlap next block
        attn_block<false>(&k_lds[kb & 1][0], &v_lds[kb & 1][0], p_lds,
                          qa, oacc, mrow, lrow, kb * KT, q0, half, ln, wave);
    }
    asm volatile("s_wait_asynccnt 0x0" ::: "memory");
    __syncthreads();
    attn_block<true>(&k_lds[kb & 1][0], &v_lds[kb & 1][0], p_lds,
                     qa, oacc, mrow, lrow, kb * KT, q0, half, ln, wave);

    // --- normalize and write out ----------------------------------------------
#pragma unroll
    for (int vv = 0; vv < 8; ++vv) {
        const int M = vv + (half ? 8 : 0);
        const float inv = 1.0f / lrow[vv];
        float* orow = out + ((size_t)(bidx * Sc + q0 + M) * (Hc * Dc)) + h * Dc;
#pragma unroll
        for (int t = 0; t < 8; ++t) orow[t * 16 + ln] = oacc[t][vv] * inv;
    }
}

// ---------------------------------------------------------------------------
// Fallback (ws too small): self-converting variant, synchronous staging.
// ---------------------------------------------------------------------------
__global__ __launch_bounds__(128)
void attn_fwd_f32src(const float* __restrict__ q,
                     const float* __restrict__ k,
                     const float* __restrict__ v,
                     float* __restrict__ out) {
    __shared__ unsigned short k_lds[KT * K_STRIDE];
    __shared__ unsigned short v_lds[Dc * V_STRIDE];
    __shared__ unsigned short p_lds[4 * QT * P_STRIDE];

    const int tid  = threadIdx.x;
    const int wave = tid >> 5;
    const int lane = tid & 31;
    const int half = lane >> 4;
    const int ln   = lane & 15;

    const int wg    = blockIdx.x;
    const int qtile = wg % (Sc / QT);
    const int kvh   = (wg / (Sc / QT)) % HKc;
    const int bidx  = wg / ((Sc / QT) * HKc);
    const int q0    = qtile * QT;
    const int h     = kvh * REP + wave;

    Frag qa[4];
    {
        const float* qrow = q + (((size_t)(bidx * Sc + q0 + ln) * Hc + h) * Dc);
#pragma unroll
        for (int c = 0; c < 4; ++c) {
#pragma unroll
            for (int vv = 0; vv < 8; ++vv) {
                int d = c * 32 + 2 * vv + (vv >= 4 ? 8 : 0) + (half ? 8 : 0);
                float2 f = *(const float2*)(qrow + d);
                qa[c].u[vv] = pack2bf(f.x * SCALE_LOG2E, f.y * SCALE_LOG2E);
            }
        }
    }

    v8f oacc[8];
#pragma unroll
    for (int t = 0; t < 8; ++t) oacc[t] = vzero8();
    float mrow[8], lrow[8];
#pragma unroll
    for (int vv = 0; vv < 8; ++vv) { mrow[vv] = -1e30f; lrow[vv] = 0.0f; }

    const int nkb = (q0 >> 5) + 1;

    for (int kb = 0; kb < nkb; ++kb) {
        const int kstart = kb * KT;
        __syncthreads();
#pragma unroll
        for (int i = 0; i < 8; ++i) {
            int f  = i * 128 + tid;
            int r  = f >> 5;
            int c4 = f & 31;
            size_t gidx = (((size_t)(bidx * Sc + kstart + r) * HKc + kvh) * Dc) + c4 * 4;
            float4 kd = *(const float4*)(k + gidx);
            *(unsigned int*)(&k_lds[r * K_STRIDE + c4 * 4])     = pack2bf(kd.x, kd.y);
            *(unsigned int*)(&k_lds[r * K_STRIDE + c4 * 4 + 2]) = pack2bf(kd.z, kd.w);
            float4 vd = *(const float4*)(v + gidx);
            v_lds[(c4 * 4 + 0) * V_STRIDE + r] = f2bf(vd.x);
            v_lds[(c4 * 4 + 1) * V_STRIDE + r] = f2bf(vd.y);
            v_lds[(c4 * 4 + 2) * V_STRIDE + r] = f2bf(vd.z);
            v_lds[(c4 * 4 + 3) * V_STRIDE + r] = f2bf(vd.w);
        }
        __syncthreads();

        if (kb == nkb - 1)
            attn_block<true>(k_lds, v_lds, p_lds, qa, oacc, mrow, lrow,
                             kstart, q0, half, ln, wave);
        else
            attn_block<false>(k_lds, v_lds, p_lds, qa, oacc, mrow, lrow,
                              kstart, q0, half, ln, wave);
    }

#pragma unroll
    for (int vv = 0; vv < 8; ++vv) {
        const int M = vv + (half ? 8 : 0);
        const float inv = 1.0f / lrow[vv];
        float* orow = out + ((size_t)(bidx * Sc + q0 + M) * (Hc * Dc)) + h * Dc;
#pragma unroll
        for (int t = 0; t < 8; ++t) orow[t * 16 + ln] = oacc[t][vv] * inv;
    }
}

extern "C" void kernel_launch(void* const* d_in, const int* in_sizes, int n_in,
                              void* d_out, int out_size, void* d_ws, size_t ws_size,
                              hipStream_t stream) {
    const float* q = (const float*)d_in[0];
    const float* k = (const float*)d_in[1];
    const float* v = (const float*)d_in[2];
    float* out = (float*)d_out;

    const size_t kv_elems = (size_t)Bc * HKc * Sc * Dc;            // 4.19M per tensor
    const size_t need     = kv_elems * 2 * sizeof(unsigned short); // 16 MB

    dim3 grid(Bc * HKc * (Sc / QT));   // 2048 workgroups
    dim3 block(128);                   // 4 waves: one query head each

    if (ws_size >= need) {
        unsigned short* kbf = (unsigned short*)d_ws;
        unsigned short* vbf = kbf + kv_elems;
        kv_to_bf16<<<dim3(Bc * HKc * (Sc / 64)), dim3(256), 0, stream>>>(k, v, kbf, vbf);
        attn_fwd_bf16ws<<<grid, block, 0, stream>>>(kbf, vbf, q, out);
    } else {
        attn_fwd_f32src<<<grid, block, 0, stream>>>(q, k, v, out);
    }
}
